// Seq2Seq_59098749993564
// MI455X (gfx1250) — matrix operand order repaired and checked by reading, not compiled, not run on hardware
//
#include <hip/hip_runtime.h>
#include <hip/hip_bf16.h>

#define MAX_LEN 64
#define BATCH   128
#define HID     1024
#define THREE_H 3072
#define VOCAB   8192

typedef __attribute__((ext_vector_type(16))) _Float16 v16h;
typedef __attribute__((ext_vector_type(8)))  float    v8f;

union Frag16 { v16h v; uint4 u[2]; };

// A fragment: 16x32 f16, row-major source (rows m, leading dim lda).
// ISA layout: lanes 0-15 M=0..15 hold K {0..7,16..23}; lanes 16-31 hold K {8..15,24..31}.
__device__ __forceinline__ v16h ld_fragA(const _Float16* __restrict__ base, int lda,
                                         int lane, int k0) {
  const int m  = lane & 15;
  const int kb = k0 + ((lane >> 4) << 3);
  const _Float16* p = base + (size_t)m * lda + kb;
  Frag16 f;
  f.u[0] = *(const uint4*)(p);       // K = kb .. kb+7
  f.u[1] = *(const uint4*)(p + 16);  // K = kb+16 .. kb+23
  return f.v;
}

// B fragment: 32x16 f16 where B[k][n] = W[n][k], W row-major [N, K] (ldk).
// Lane n = lane&15; lanes 0-15 hold K {0..7,16..23}; lanes 16-31 hold K {8..15,24..31}.
__device__ __forceinline__ v16h ld_fragB(const _Float16* __restrict__ w, int ldk,
                                         int lane, int k0) {
  const int n  = lane & 15;
  const int kb = k0 + ((lane >> 4) << 3);
  const _Float16* p = w + (size_t)n * ldk + kb;
  Frag16 f;
  f.u[0] = *(const uint4*)(p);
  f.u[1] = *(const uint4*)(p + 16);
  return f.v;
}

__device__ __forceinline__ v8f wmma32(v16h a, v16h b, v8f c) {
  return __builtin_amdgcn_wmma_f32_16x16x32_f16(false, a, false, b, (short)0, c,
                                                false, false);
}

// Fused GRU layer. HAS_IH=true: layer1, input gates via WMMA against w_ih (f16).
// HAS_IH=false: layer0, input is one-hot(tok) so the input-gate "GEMM" is a
// direct f32 column gather of dec_w_ih0 fused into the epilogue (full precision,
// no intermediate buffer, no extra kernel launch).
// Block = 128 threads (4 wave32s = 4 M-tiles); blockIdx.y picks batch half.
template <bool HAS_IH>
__global__ __launch_bounds__(128) void gru_layer_wmma(
    const _Float16* __restrict__ x_h,    // [128,1024] layer input f16 (layer1) or null
    const float*    __restrict__ w_ih0,  // [3072,8192] f32 (layer0) or null
    const float*    __restrict__ b_ih0,  // [3072] (layer0) or null
    const int*      __restrict__ tok,    // [128] (layer0) or null
    const _Float16* __restrict__ h_h,    // [128,1024] prev hidden f16
    const float*    __restrict__ h_f,    // [128,1024] prev hidden f32
    const _Float16* __restrict__ w_ih,   // [3072,1024] f16 (layer1) or null
    const _Float16* __restrict__ w_hh,   // [3072,1024] f16
    const float*    __restrict__ b_ih,   // [3072] (layer1) or null
    const float*    __restrict__ b_hh,   // [3072]
    float*          __restrict__ hn_f,   // out [128,1024] f32
    _Float16*       __restrict__ hn_h)   // out [128,1024] f16
{
  const int lane = threadIdx.x & 31;
  const int mt   = (blockIdx.y << 2) + (threadIdx.x >> 5);  // 8 M-tiles over grid.y
  const int n0   = blockIdx.x * 16;                         // 16-col N-tile in HID

  const _Float16* hrow = h_h + (size_t)mt * 16 * HID;
  const _Float16* whr  = w_hh + (size_t)(0 * HID + n0) * HID;
  const _Float16* whz  = w_hh + (size_t)(1 * HID + n0) * HID;
  const _Float16* whn  = w_hh + (size_t)(2 * HID + n0) * HID;

  const _Float16* xrow = nullptr;
  const _Float16 *wir = nullptr, *wiz = nullptr, *win = nullptr;
  if (HAS_IH) {
    xrow = x_h + (size_t)mt * 16 * HID;
    wir = w_ih + (size_t)(0 * HID + n0) * HID;
    wiz = w_ih + (size_t)(1 * HID + n0) * HID;
    win = w_ih + (size_t)(2 * HID + n0) * HID;
  }

  v8f ahr = {0, 0, 0, 0, 0, 0, 0, 0}, ahz = ahr, ahn = ahr;
  v8f air = ahr, aiz = ahr, ain = ahr;

#pragma unroll 2
  for (int k0 = 0; k0 < HID; k0 += 32) {
    v16h ah = ld_fragA(hrow, HID, lane, k0);
    ahr = wmma32(ah, ld_fragB(whr, HID, lane, k0), ahr);
    ahz = wmma32(ah, ld_fragB(whz, HID, lane, k0), ahz);
    ahn = wmma32(ah, ld_fragB(whn, HID, lane, k0), ahn);
    if (HAS_IH) {
      v16h ax = ld_fragA(xrow, HID, lane, k0);
      air = wmma32(ax, ld_fragB(wir, HID, lane, k0), air);
      aiz = wmma32(ax, ld_fragB(wiz, HID, lane, k0), aiz);
      ain = wmma32(ax, ld_fragB(win, HID, lane, k0), ain);
    }
  }

  // D layout: acc element v -> row mt*16 + v + 8*(lane>>4), col n0 + (lane&15)
  const int n     = n0 + (lane & 15);
  const int mbase = mt * 16 + ((lane >> 4) << 3);
  const float bhr = b_hh[n], bhz = b_hh[HID + n], bhn = b_hh[2 * HID + n];
  float bir, biz, bin_;
  if (HAS_IH) {
    bir = b_ih[n]; biz = b_ih[HID + n]; bin_ = b_ih[2 * HID + n];
  } else {
    bir = b_ih0[n]; biz = b_ih0[HID + n]; bin_ = b_ih0[2 * HID + n];
  }

#pragma unroll
  for (int v = 0; v < 8; ++v) {
    const int m = mbase + v;
    float ir, iz, inn;
    if (HAS_IH) {
      ir = air[v] + bir; iz = aiz[v] + biz; inn = ain[v] + bin_;
    } else {
      const int tm = tok[m];  // one-hot input GEMM == column gather (f32)
      ir  = w_ih0[(size_t)(0 * HID + n) * VOCAB + tm] + bir;
      iz  = w_ih0[(size_t)(1 * HID + n) * VOCAB + tm] + biz;
      inn = w_ih0[(size_t)(2 * HID + n) * VOCAB + tm] + bin_;
    }
    const float hr = ahr[v] + bhr, hz = ahz[v] + bhz, hnv = ahn[v] + bhn;
    const float r  = 1.f / (1.f + __expf(-(ir + hr)));
    const float z  = 1.f / (1.f + __expf(-(iz + hz)));
    const float nn = tanhf(inn + r * hnv);
    const float hp = h_f[(size_t)m * HID + n];
    const float hw = (1.f - z) * nn + z * hp;
    hn_f[(size_t)m * HID + n] = hw;
    hn_h[(size_t)m * HID + n] = (_Float16)hw;
  }
}

// logits = h1 @ fc_w^T + fc_b  (128 x 8192, K=1024). 1024 WGs x 4 waves.
__global__ __launch_bounds__(128) void logits_wmma(
    const _Float16* __restrict__ h_h, const _Float16* __restrict__ fcw,
    const float* __restrict__ fcb, float* __restrict__ logits)
{
  const int lane = threadIdx.x & 31;
  const int mt   = (blockIdx.y << 2) + (threadIdx.x >> 5);
  const int n0   = blockIdx.x * 16;
  const _Float16* hrow = h_h + (size_t)mt * 16 * HID;
  const _Float16* wrow = fcw + (size_t)n0 * HID;
  v8f acc = {0, 0, 0, 0, 0, 0, 0, 0};
#pragma unroll 4
  for (int k0 = 0; k0 < HID; k0 += 32)
    acc = wmma32(ld_fragA(hrow, HID, lane, k0), ld_fragB(wrow, HID, lane, k0), acc);
  const int n     = n0 + (lane & 15);
  const int mbase = mt * 16 + ((lane >> 4) << 3);
  const float bb  = fcb[n];
#pragma unroll
  for (int v = 0; v < 8; ++v)
    logits[(size_t)(mbase + v) * VOCAB + n] = acc[v] + bb;
}

// Per-row log_softmax + argmax + teacher-forced token select.
__global__ __launch_bounds__(256) void softmax_argmax_k(
    const float* __restrict__ logits, const int* __restrict__ oseq_t,
    const unsigned char* __restrict__ tf_t, float* __restrict__ out_t,
    int* __restrict__ tok)
{
  __shared__ float sred[256];
  __shared__ int   sidx[256];
  const int b = blockIdx.x, tid = threadIdx.x;
  const float* row = logits + (size_t)b * VOCAB;

  float mx = -3.402823466e38f; int mi = 0;
  for (int v = tid; v < VOCAB; v += 256) {
    const float x = row[v];
    if (x > mx) { mx = x; mi = v; }
  }
  sred[tid] = mx; sidx[tid] = mi;
  __syncthreads();
  for (int s = 128; s; s >>= 1) {
    if (tid < s) {
      const float o = sred[tid + s]; const int oi = sidx[tid + s];
      if (o > sred[tid] || (o == sred[tid] && oi < sidx[tid])) {
        sred[tid] = o; sidx[tid] = oi;
      }
    }
    __syncthreads();
  }
  const float rmax = sred[0]; const int amax = sidx[0];
  __syncthreads();

  float s = 0.f;
  for (int v = tid; v < VOCAB; v += 256) s += __expf(row[v] - rmax);
  sred[tid] = s;
  __syncthreads();
  for (int st = 128; st; st >>= 1) {
    if (tid < st) sred[tid] += sred[tid + st];
    __syncthreads();
  }
  const float lse = __logf(sred[0]);

  for (int v = tid; v < VOCAB; v += 256)
    out_t[(size_t)b * VOCAB + v] = row[v] - rmax - lse;
  if (tid == 0) tok[b] = tf_t[0] ? oseq_t[b] : amax;
}

__global__ void f32_to_f16_k(const float* __restrict__ s, _Float16* __restrict__ d, int n) {
  for (int i = blockIdx.x * 256 + threadIdx.x; i < n; i += gridDim.x * 256)
    d[i] = (_Float16)s[i];
}
__global__ void zero_f32_k(float* p, int n) {
  for (int i = blockIdx.x * 256 + threadIdx.x; i < n; i += gridDim.x * 256) p[i] = 0.f;
}
__global__ void zero_f16_k(_Float16* p, int n) {
  for (int i = blockIdx.x * 256 + threadIdx.x; i < n; i += gridDim.x * 256)
    p[i] = (_Float16)0.f;
}
__global__ void init_tok_k(int* tok) { tok[threadIdx.x] = 0; }

extern "C" void kernel_launch(void* const* d_in, const int* in_sizes, int n_in,
                              void* d_out, int out_size, void* d_ws, size_t ws_size,
                              hipStream_t stream) {
  (void)in_sizes; (void)n_in; (void)out_size; (void)ws_size;
  // Encoder inputs (0, 3..10) are dead: reference decoder starts from zero hidden.
  const int*           output_seq = (const int*)d_in[1];
  const unsigned char* tf         = (const unsigned char*)d_in[2];  // numpy bool
  const float* w_ih0 = (const float*)d_in[11];
  const float* w_hh0 = (const float*)d_in[12];
  const float* b_ih0 = (const float*)d_in[13];
  const float* b_hh0 = (const float*)d_in[14];
  const float* w_ih1 = (const float*)d_in[15];
  const float* w_hh1 = (const float*)d_in[16];
  const float* b_ih1 = (const float*)d_in[17];
  const float* b_hh1 = (const float*)d_in[18];
  const float* fc_w  = (const float*)d_in[19];
  const float* fc_b  = (const float*)d_in[20];
  float* out = (float*)d_out;

  char* ws = (char*)d_ws;
  size_t off = 0;
  auto carve = [&](size_t bytes) -> char* {
    char* p = ws + off;
    off = (off + bytes + 255) & ~(size_t)255;
    return p;
  };
  _Float16* whh0h = (_Float16*)carve((size_t)THREE_H * HID * 2);
  _Float16* wih1h = (_Float16*)carve((size_t)THREE_H * HID * 2);
  _Float16* whh1h = (_Float16*)carve((size_t)THREE_H * HID * 2);
  _Float16* fcwh  = (_Float16*)carve((size_t)VOCAB * HID * 2);
  float*    h0f[2] = {(float*)carve(BATCH * HID * 4), (float*)carve(BATCH * HID * 4)};
  _Float16* h0h[2] = {(_Float16*)carve(BATCH * HID * 2), (_Float16*)carve(BATCH * HID * 2)};
  float*    h1f[2] = {(float*)carve(BATCH * HID * 4), (float*)carve(BATCH * HID * 4)};
  _Float16* h1h[2] = {(_Float16*)carve(BATCH * HID * 2), (_Float16*)carve(BATCH * HID * 2)};
  float* logits = (float*)carve((size_t)BATCH * VOCAB * 4);
  int*   tok    = (int*)carve(512);

  // One-time per launch: f16 weight images + zero initial state + out[0]=0.
  f32_to_f16_k<<<2048, 256, 0, stream>>>(w_hh0, whh0h, THREE_H * HID);
  f32_to_f16_k<<<2048, 256, 0, stream>>>(w_ih1, wih1h, THREE_H * HID);
  f32_to_f16_k<<<2048, 256, 0, stream>>>(w_hh1, whh1h, THREE_H * HID);
  f32_to_f16_k<<<4096, 256, 0, stream>>>(fc_w, fcwh, VOCAB * HID);
  zero_f32_k<<<2048, 256, 0, stream>>>(out, BATCH * VOCAB);  // dec_outputs[0]
  zero_f32_k<<<512, 256, 0, stream>>>(h0f[0], BATCH * HID);
  zero_f32_k<<<512, 256, 0, stream>>>(h1f[0], BATCH * HID);
  zero_f16_k<<<512, 256, 0, stream>>>(h0h[0], BATCH * HID);
  zero_f16_k<<<512, 256, 0, stream>>>(h1h[0], BATCH * HID);
  init_tok_k<<<1, 128, 0, stream>>>(tok);

  const dim3 gruGrid(HID / 16, 2);     // 128 WGs x 4 waves
  const dim3 fcGrid(VOCAB / 16, 2);    // 1024 WGs x 4 waves

  for (int t = 1; t < MAX_LEN; ++t) {
    const int cur = (t + 1) & 1, nxt = t & 1;
    gru_layer_wmma<false><<<gruGrid, 128, 0, stream>>>(
        nullptr, w_ih0, b_ih0, tok, h0h[cur], h0f[cur], nullptr, whh0h,
        nullptr, b_hh0, h0f[nxt], h0h[nxt]);
    gru_layer_wmma<true><<<gruGrid, 128, 0, stream>>>(
        h0h[nxt], nullptr, nullptr, nullptr, h1h[cur], h1f[cur], wih1h, whh1h,
        b_ih1, b_hh1, h1f[nxt], h1h[nxt]);
    logits_wmma<<<fcGrid, 128, 0, stream>>>(h1h[nxt], fcwh, fc_b, logits);
    softmax_argmax_k<<<BATCH, 256, 0, stream>>>(
        logits, output_seq + t * BATCH, tf + t, out + (size_t)t * BATCH * VOCAB, tok);
  }
}
